// CrossAttentionAudio2T_49847390437708
// MI455X (gfx1250) — matrix operand order repaired and checked by reading, not compiled, not run on hardware
//
#include <hip/hip_runtime.h>

// ---------------- types ----------------
typedef __attribute__((ext_vector_type(16))) __bf16 v16bf;
typedef __attribute__((ext_vector_type(8)))  __bf16 v8bf;
typedef __attribute__((ext_vector_type(8)))  float  v8f;

// ---------------- problem constants ----------------
constexpr int Bc = 4, Tc = 8, Nc = 196, NAc = 196, STc = 8;
constexpr int Dm = 768, Hh = 12, HDd = 64;
constexpr int LQ = Tc * Nc;   // 1568
constexpr int LK = NAc * STc; // 1568
constexpr float SCL = 0.125f; // 64^-0.5
constexpr int KSTEPS = Dm / 32; // 24

// ---------------- WMMA helpers ----------------
static __device__ __forceinline__ v16bf mk16(v8bf lo, v8bf hi) {
  v16bf r;
#pragma unroll
  for (int i = 0; i < 8; ++i) { r[i] = lo[i]; r[i + 8] = hi[i]; }
  return r;
}

// A-operand (16x32 bf16): lane holds M=lane16; element e -> K = (e&7) + (e>>3)*16 + hg*8
static __device__ __forceinline__ v16bf loadA_g(const __bf16* base, int lane16, int hg, int ld) {
  const __bf16* p = base + lane16 * ld + hg * 8;
  return mk16(*(const v8bf*)p, *(const v8bf*)(p + 16));
}

// B-operand (32x16 bf16): lane holds N=lane16; element e -> K = (e&7) + (e>>3)*16 + hg*8
static __device__ __forceinline__ v16bf loadB_g(const __bf16* base, int lane16, int hg, int ld) {
  const __bf16* p = base + lane16 * ld + hg * 8;
  return mk16(*(const v8bf*)p, *(const v8bf*)(p + 16));
}

static __device__ __forceinline__ v8f wmma_bf16(v16bf a, v16bf b, v8f c) {
  return __builtin_amdgcn_wmma_f32_16x16x32_bf16(false, a, false, b, (short)0, c, false, false);
}

static __device__ __forceinline__ float rmax16(float v) {
#pragma unroll
  for (int m = 1; m < 16; m <<= 1) v = fmaxf(v, __shfl_xor(v, m, 16));
  return v;
}
static __device__ __forceinline__ float rsum16(float v) {
#pragma unroll
  for (int m = 1; m < 16; m <<= 1) v += __shfl_xor(v, m, 16);
  return v;
}

// ---------------- prep kernels ----------------
__global__ void cast_bf16_kernel(const float* __restrict__ s, __bf16* __restrict__ d, int n) {
  int i = blockIdx.x * blockDim.x + threadIdx.x;
  if (i < n) d[i] = (__bf16)s[i];
}

// X[b][(t*N+n)][d] = t_x + vmae_space_pos[n][d] + vmae_temporal_pos[t][d]
__global__ void prep_x_kernel(const float* __restrict__ t_x, const float* __restrict__ vsp,
                              const float* __restrict__ vtp, __bf16* __restrict__ X) {
  int idx = blockIdx.x * blockDim.x + threadIdx.x;
  if (idx >= Bc * LQ * Dm) return;
  int d = idx % Dm;
  int rest = idx / Dm;
  int n = rest % Nc;
  int t = (rest / Nc) % Tc;
  X[idx] = (__bf16)(t_x[idx] + vsp[n * Dm + d] + vtp[t * Dm + d]);
}

// A[b][(n*ST+t)][d] = audio[2+n][b*ST+t][d] + audio_space_pos[n][d] + audio_temporal_pos[t][d]
__global__ void prep_a_kernel(const float* __restrict__ audio, const float* __restrict__ asp,
                              const float* __restrict__ atp, __bf16* __restrict__ A) {
  int idx = blockIdx.x * blockDim.x + threadIdx.x;
  if (idx >= Bc * LK * Dm) return;
  int d = idx % Dm;
  int rest = idx / Dm;
  int t = rest % STc;
  int n = (rest / STc) % NAc;
  int b = rest / (STc * NAc);
  float v = audio[((size_t)(2 + n) * (Bc * STc) + (b * STc + t)) * Dm + d]
          + asp[n * Dm + d] + atp[t * Dm + d];
  A[idx] = (__bf16)v;
}

// ---------------- pipelined 32x64 GEMM body (ping-pong, fully unrolled) ----------------
template <int LDA>
static __device__ __forceinline__ void gemm32x64_body(const __bf16* __restrict__ Arow0,
                                                      const __bf16* __restrict__ Arow1,
                                                      const __bf16* __restrict__ W,
                                                      int cBase, int lane16, int hg,
                                                      v8f acc[2][4]) {
  v16bf A[2][2];
  v16bf Bm[2][4];

  auto loadStep = [&](int slot, int k) {
    A[slot][0] = loadA_g(Arow0 + k, lane16, hg, LDA);
    A[slot][1] = loadA_g(Arow1 + k, lane16, hg, LDA);
#pragma unroll
    for (int c = 0; c < 4; ++c)
      Bm[slot][c] = loadB_g(W + (size_t)(cBase + c * 16) * Dm + k, lane16, hg, Dm);
  };
  auto compute = [&](int slot) {
#pragma unroll
    for (int c = 0; c < 4; ++c) {
      acc[0][c] = wmma_bf16(A[slot][0], Bm[slot][c], acc[0][c]);
      acc[1][c] = wmma_bf16(A[slot][1], Bm[slot][c], acc[1][c]);
    }
  };

  loadStep(0, 0);
#pragma unroll
  for (int s = 0; s + 2 < KSTEPS; s += 2) {
    loadStep(1, (s + 1) * 32);
    compute(0);
    loadStep(0, (s + 2) * 32);
    compute(1);
  }
  loadStep(1, (KSTEPS - 1) * 32);
  compute(0);
  compute(1);
}

// ---------------- Q projection: Q[b][h][lq][hd] = SCL*(X @ q_w^T + q_b) ----------------
__global__ __launch_bounds__(32, 2) void gemm_q_kernel(const __bf16* __restrict__ X,
                                                       const __bf16* __restrict__ W,
                                                       const float* __restrict__ bias,
                                                       __bf16* __restrict__ Q) {
  int lane = threadIdx.x, lane16 = lane & 15, hg = lane >> 4;
  int rowBase = blockIdx.x * 32;          // over B*LQ
  int cBase = blockIdx.y * 64;            // over D
  v8f acc[2][4] = {};
  gemm32x64_body<Dm>(X + (size_t)rowBase * Dm, X + (size_t)(rowBase + 16) * Dm,
                     W, cBase, lane16, hg, acc);
  int b_ = rowBase / LQ;
#pragma unroll
  for (int c = 0; c < 4; ++c) {
    int col = cBase + c * 16 + lane16;
    float bv = bias[col];
    int h = col >> 6, hd = col & 63;
#pragma unroll
    for (int t = 0; t < 2; ++t) {
      int lqBase = rowBase % LQ + t * 16;
#pragma unroll
      for (int r = 0; r < 8; ++r) {
        int lq = lqBase + r + 8 * hg;
        Q[(((size_t)(b_ * Hh + h) * LQ + lq) << 6) + hd] = (__bf16)((acc[t][c][r] + bv) * SCL);
      }
    }
  }
}

// ---------------- KV projection: K[b][h][lk][hd], VT[b][h][hd][lk] ----------------
__global__ __launch_bounds__(32, 2) void gemm_kv_kernel(const __bf16* __restrict__ A,
                                                        const __bf16* __restrict__ W,
                                                        const float* __restrict__ bias,
                                                        __bf16* __restrict__ Kb,
                                                        __bf16* __restrict__ VT) {
  int lane = threadIdx.x, lane16 = lane & 15, hg = lane >> 4;
  int rowBase = blockIdx.x * 32;          // over B*LK
  int cBase = blockIdx.y * 64;            // over 2*D
  v8f acc[2][4] = {};
  gemm32x64_body<Dm>(A + (size_t)rowBase * Dm, A + (size_t)(rowBase + 16) * Dm,
                     W, cBase, lane16, hg, acc);
  int b_ = rowBase / LK;
#pragma unroll
  for (int c = 0; c < 4; ++c) {
    int col0 = cBase + c * 16;
    float bv = bias[col0 + lane16];
    int which = (col0 >= Dm) ? 1 : 0;
    int f = col0 - which * Dm;
    int h = f >> 6;
    int hd = (f & 63) + lane16;
#pragma unroll
    for (int t = 0; t < 2; ++t) {
      int lkBase = rowBase % LK + t * 16;
#pragma unroll
      for (int r = 0; r < 8; ++r) {
        int lk = lkBase + r + 8 * hg;
        __bf16 v = (__bf16)(acc[t][c][r] + bv);
        if (!which)
          Kb[(((size_t)(b_ * Hh + h) * LK + lk) << 6) + hd] = v;
        else
          VT[((size_t)(b_ * Hh + h) * HDd + hd) * LK + lk] = v;
      }
    }
  }
}

// ---------------- flash attention: per (b*h, 32-query tile), ping-pong K/V tiles ----------------
__global__ __launch_bounds__(32, 2) void attn_kernel(const __bf16* __restrict__ Q,
                                                     const __bf16* __restrict__ Kb,
                                                     const __bf16* __restrict__ VT,
                                                     __bf16* __restrict__ Ob) {
  constexpr int LDSTRIDE = 40; // bf16 elems, padded (80B rows, 16B-aligned chunks)
  __shared__ __bf16 pl[32 * LDSTRIDE];
  int lane = threadIdx.x, lane16 = lane & 15, hg = lane >> 4;
  int bh = blockIdx.x;        // 0..B*H-1
  int qt = blockIdx.y;        // 0..LQ/32-1
  const __bf16* q = Q + ((size_t)bh * LQ + qt * 32) * HDd;
  const __bf16* k = Kb + (size_t)bh * LK * HDd;
  const __bf16* vt = VT + (size_t)bh * HDd * LK;

  // 32-query tile: two 16-row A fragments per HD half
  v16bf qa[2][2];
#pragma unroll
  for (int t = 0; t < 2; ++t) {
    qa[t][0] = loadA_g(q + (size_t)t * 16 * HDd + 0, lane16, hg, HDd);
    qa[t][1] = loadA_g(q + (size_t)t * 16 * HDd + 32, lane16, hg, HDd);
  }

  float m[2][8], l[2][8];
#pragma unroll
  for (int t = 0; t < 2; ++t)
#pragma unroll
    for (int r = 0; r < 8; ++r) { m[t][r] = -3.0e38f; l[t][r] = 0.f; }
  v8f Oc[2][4] = {};

  v16bf kt[2][4];   // [slot][g*2+s]: keys (kb+g*16), hd half s
  v16bf vtl[2][4];  // [slot][c]: VT rows c*16.., keys kb..kb+31

  auto loadKV = [&](int slot, int kb) {
    kt[slot][0] = loadB_g(k + (size_t)kb * HDd + 0, lane16, hg, HDd);
    kt[slot][1] = loadB_g(k + (size_t)kb * HDd + 32, lane16, hg, HDd);
    kt[slot][2] = loadB_g(k + (size_t)(kb + 16) * HDd + 0, lane16, hg, HDd);
    kt[slot][3] = loadB_g(k + (size_t)(kb + 16) * HDd + 32, lane16, hg, HDd);
#pragma unroll
    for (int c = 0; c < 4; ++c)
      vtl[slot][c] = loadB_g(vt + (size_t)(c * 16) * LK + kb, lane16, hg, LK);
  };

  auto process = [&](int slot) {
    v8f S[2][2];
#pragma unroll
    for (int t = 0; t < 2; ++t) {
      S[t][0] = v8f{};
      S[t][0] = wmma_bf16(qa[t][0], kt[slot][0], S[t][0]);
      S[t][0] = wmma_bf16(qa[t][1], kt[slot][1], S[t][0]);
      S[t][1] = v8f{};
      S[t][1] = wmma_bf16(qa[t][0], kt[slot][2], S[t][1]);
      S[t][1] = wmma_bf16(qa[t][1], kt[slot][3], S[t][1]);
    }

    // online softmax per row (row = r + 8*hg within each 16-row sub-tile)
#pragma unroll
    for (int t = 0; t < 2; ++t) {
#pragma unroll
      for (int r = 0; r < 8; ++r) {
        float smax = rmax16(fmaxf(S[t][0][r], S[t][1][r]));
        float mnew = fmaxf(m[t][r], smax);
        float alpha = __expf(m[t][r] - mnew);
        float p0 = __expf(S[t][0][r] - mnew);
        float p1 = __expf(S[t][1][r] - mnew);
        float rsum = rsum16(p0 + p1);
        l[t][r] = l[t][r] * alpha + rsum;
        m[t][r] = mnew;
#pragma unroll
        for (int c = 0; c < 4; ++c) Oc[t][c][r] *= alpha;
        int row = t * 16 + r + 8 * hg;
        pl[row * LDSTRIDE + lane16] = (__bf16)p0;
        pl[row * LDSTRIDE + 16 + lane16] = (__bf16)p1;
      }
    }
    __syncthreads();

    // reload P in A-layout (C->A transpose via LDS)
    v16bf Pa[2];
#pragma unroll
    for (int t = 0; t < 2; ++t) {
      const __bf16* pp = &pl[(t * 16 + lane16) * LDSTRIDE + hg * 8];
      Pa[t] = mk16(*(const v8bf*)pp, *(const v8bf*)(pp + 16));
    }

    // O += P @ V (V tiles already resident from the ping-pong prefetch)
#pragma unroll
    for (int c = 0; c < 4; ++c) {
      Oc[0][c] = wmma_bf16(Pa[0], vtl[slot][c], Oc[0][c]);
      Oc[1][c] = wmma_bf16(Pa[1], vtl[slot][c], Oc[1][c]);
    }
    __syncthreads();
  };

  loadKV(0, 0);
  int kb = 0;
  for (; kb + 32 < LK; kb += 64) {
    // cache-level prefetch one pair further ahead (global_prefetch_b8)
    if (kb + 96 < LK) {
      __builtin_prefetch((const void*)(k + (size_t)(kb + 96 + lane) * HDd), 0, 1);
      __builtin_prefetch((const void*)(vt + (size_t)(lane * 2) * LK + kb + 96), 0, 1);
    }
    loadKV(1, kb + 32);
    process(0);
    loadKV(0, kb + 64);  // kb <= LK-96 inside this loop, so kb+64 is always valid
    process(1);
  }
  // one remaining step (LK/32 is odd): slot0 holds keys [LK-32, LK)
  process(0);

  // normalize and scatter to O buffer [B*LQ][D], col = h*64 + hd
  int b_ = bh / Hh, h = bh % Hh;
#pragma unroll
  for (int t = 0; t < 2; ++t) {
#pragma unroll
    for (int r = 0; r < 8; ++r) {
      float inv = 1.0f / l[t][r];
      int row = b_ * LQ + qt * 32 + t * 16 + r + 8 * hg;
#pragma unroll
      for (int c = 0; c < 4; ++c) {
        int col = h * 64 + c * 16 + lane16;
        Ob[(size_t)row * Dm + col] = (__bf16)(Oc[t][c][r] * inv);
      }
    }
  }
}

// ---------------- output projection: out = O @ proj_w^T + proj_b (f32) ----------------
__global__ __launch_bounds__(32, 2) void gemm_proj_kernel(const __bf16* __restrict__ O,
                                                          const __bf16* __restrict__ W,
                                                          const float* __restrict__ bias,
                                                          float* __restrict__ out) {
  int lane = threadIdx.x, lane16 = lane & 15, hg = lane >> 4;
  int rowBase = blockIdx.x * 32;
  int cBase = blockIdx.y * 64;
  v8f acc[2][4] = {};
  gemm32x64_body<Dm>(O + (size_t)rowBase * Dm, O + (size_t)(rowBase + 16) * Dm,
                     W, cBase, lane16, hg, acc);
#pragma unroll
  for (int c = 0; c < 4; ++c) {
    int col = cBase + c * 16 + lane16;
    float bv = bias[col];
#pragma unroll
    for (int t = 0; t < 2; ++t) {
#pragma unroll
      for (int r = 0; r < 8; ++r) {
        int row = rowBase + t * 16 + r + 8 * hg;
        out[(size_t)row * Dm + col] = acc[t][c][r] + bv;
      }
    }
  }
}

// ---------------- launch ----------------
extern "C" void kernel_launch(void* const* d_in, const int* in_sizes, int n_in,
                              void* d_out, int out_size, void* d_ws, size_t ws_size,
                              hipStream_t stream) {
  (void)in_sizes; (void)n_in; (void)out_size; (void)ws_size;
  const float* t_x  = (const float*)d_in[0];
  const float* audio= (const float*)d_in[1];
  const float* vsp  = (const float*)d_in[2];
  const float* asp  = (const float*)d_in[3];
  const float* vtp  = (const float*)d_in[4];
  const float* atp  = (const float*)d_in[5];
  const float* q_w  = (const float*)d_in[6];
  const float* q_b  = (const float*)d_in[7];
  const float* kv_w = (const float*)d_in[8];
  const float* kv_b = (const float*)d_in[9];
  const float* p_w  = (const float*)d_in[10];
  const float* p_b  = (const float*)d_in[11];
  float* out = (float*)d_out;

  size_t off = 0;
  auto alloc = [&](size_t bytes) -> void* {
    off = (off + 255) & ~(size_t)255;
    void* p = (char*)d_ws + off;
    off += bytes;
    return p;
  };
  const size_t ACT = (size_t)Bc * LQ * Dm;       // 4,816,896 elems
  __bf16* Xb  = (__bf16*)alloc(ACT * 2);          // prepped queries; reused as O after gemm_q
  __bf16* Ab  = (__bf16*)alloc(ACT * 2);          // prepped audio kv input
  __bf16* Qb  = (__bf16*)alloc(ACT * 2);          // [B,H,LQ,HD]
  __bf16* Kb  = (__bf16*)alloc(ACT * 2);          // [B,H,LK,HD]
  __bf16* VTb = (__bf16*)alloc(ACT * 2);          // [B,H,HD,LK]
  __bf16* qwb = (__bf16*)alloc((size_t)Dm * Dm * 2);
  __bf16* kvwb= (__bf16*)alloc((size_t)2 * Dm * Dm * 2);
  __bf16* pwb = (__bf16*)alloc((size_t)Dm * Dm * 2);
  __bf16* Ob  = Xb;   // X is dead after gemm_q; attn writes O here, proj reads it

  // weight casts
  cast_bf16_kernel<<<(Dm * Dm + 255) / 256, 256, 0, stream>>>(q_w, qwb, Dm * Dm);
  cast_bf16_kernel<<<(2 * Dm * Dm + 255) / 256, 256, 0, stream>>>(kv_w, kvwb, 2 * Dm * Dm);
  cast_bf16_kernel<<<(Dm * Dm + 255) / 256, 256, 0, stream>>>(p_w, pwb, Dm * Dm);

  // activation prep (pos embeds + rearrange, f32 -> bf16)
  prep_x_kernel<<<(int)((ACT + 255) / 256), 256, 0, stream>>>(t_x, vsp, vtp, Xb);
  prep_a_kernel<<<(int)((ACT + 255) / 256), 256, 0, stream>>>(audio, asp, atp, Ab);

  // projections (32-row x 64-col tiles per wave, ping-pong pipelined)
  gemm_q_kernel<<<dim3(Bc * LQ / 32, Dm / 64), 32, 0, stream>>>(Xb, qwb, q_b, Qb);
  gemm_kv_kernel<<<dim3(Bc * LK / 32, 2 * Dm / 64), 32, 0, stream>>>(Ab, kvwb, kv_b, Kb, VTb);

  // attention (32-query tiles)
  attn_kernel<<<dim3(Bc * Hh, LQ / 32), 32, 0, stream>>>(Qb, Kb, VTb, Ob);

  // output projection
  gemm_proj_kernel<<<dim3(Bc * LQ / 32, Dm / 64), 32, 0, stream>>>(Ob, pwb, p_b, out);
}
